// LSTM_24885040513117
// MI455X (gfx1250) — matrix operand order repaired and checked by reading, not compiled, not run on hardware
//
#include <hip/hip_runtime.h>

// LSTM: B=256, T=128, I=512, H=768, L=1.  All shapes hardcoded per reference.
#define BB 256
#define TT 128
#define II 512
#define HH 768

typedef __attribute__((ext_vector_type(16))) __bf16 v16bf;
typedef __attribute__((ext_vector_type(8)))  __bf16 v8bf;
typedef __attribute__((ext_vector_type(8)))  float  v8f;
typedef __attribute__((ext_vector_type(4)))  float  v4f;

__device__ __forceinline__ float fast_sigmoid(float x) {
    return 1.0f / (1.0f + __expf(-x));
}

__device__ __forceinline__ float fast_tanh(float x) {
#if __has_builtin(__builtin_amdgcn_tanhf)
    return __builtin_amdgcn_tanhf(x);
#elif __has_builtin(__builtin_amdgcn_tanh_f32)
    return __builtin_amdgcn_tanh_f32(x);
#else
    x = fminf(fmaxf(x, -15.0f), 15.0f);
    float e = __expf(2.0f * x);
    return (e - 1.0f) / (e + 1.0f);
#endif
}

// f32 -> bf16, 8 elements per thread (n must be a multiple of 8)
__global__ void cvt_f32_bf16_v8(const float* __restrict__ src,
                                __bf16* __restrict__ dst, int n8) {
    int i = blockIdx.x * blockDim.x + threadIdx.x;
    if (i < n8) {
        const float* s = src + (size_t)i * 8;
        v4f a = *(const v4f*)(s);
        v4f b = *(const v4f*)(s + 4);
        v8bf o;
#pragma unroll
        for (int e = 0; e < 4; ++e) {
            o[e]     = (__bf16)a[e];
            o[e + 4] = (__bf16)b[e];
        }
        *(v8bf*)(dst + (size_t)i * 8) = o;
    }
}

// h0/c0 init: state at t=0 of each sequence -> h_bf16 buf0 / c fp32
__global__ void init_state(const float* __restrict__ h0,
                           const float* __restrict__ c0,
                           __bf16* __restrict__ hbf,
                           float* __restrict__ cws) {
    int i = blockIdx.x * blockDim.x + threadIdx.x;
    if (i < BB * HH) {
        int b = i / HH, n = i % HH;
        size_t src = (size_t)b * TT * HH + n;   // h0[0, b*T + 0, n]
        hbf[i] = (__bf16)h0[src];
        cws[i] = c0[src];
    }
}

// Build a v16bf A/B fragment from two 16-byte runs (ISA 16-bit operand layout).
__device__ __forceinline__ v16bf frag16(const __bf16* p0, const __bf16* p1) {
    v8bf lo = *(const v8bf*)p0;
    v8bf hi = *(const v8bf*)p1;
    return __builtin_shufflevector(lo, hi,
        0, 1, 2, 3, 4, 5, 6, 7, 8, 9, 10, 11, 12, 13, 14, 15);
}

// One LSTM timestep, fused: g = x_t @ Wih^T + h @ Whh^T + b, gates, c/h update.
// Wave tile: M=32 (2 A-subtiles), N=16 hidden units x 4 gates -> 8 WMMA accums.
__global__ __launch_bounds__(256)
void lstm_step(const __bf16* __restrict__ xbf,         // [B*T, I] bf16
               const unsigned char* __restrict__ term, // [B*T] bool
               const __bf16* __restrict__ wih,         // [4H, I] bf16
               const __bf16* __restrict__ whh,         // [4H, H] bf16
               const float* __restrict__ bih,          // [4H]
               const float* __restrict__ bhh,          // [4H]
               const __bf16* __restrict__ h_in,        // [B, H] bf16
               __bf16* __restrict__ h_out,             // [B, H] bf16
               float* __restrict__ c,                  // [B, H] f32 (in/out)
               float* __restrict__ out,                // [B*T, H] f32
               float* __restrict__ hf,                 // [B, H] (written at t=T-1)
               float* __restrict__ cf,                 // [B, H] (written at t=T-1)
               int t)
{
    const int lane   = threadIdx.x & 31;
    const int wave   = threadIdx.x >> 5;
    const int job    = blockIdx.x * 8 + wave;   // 0..383
    const int n_tile = job % (HH / 16);         // 0..47
    const int m32    = job / (HH / 16);         // 0..7
    const int m_base = m32 * 32;
    const int n_base = n_tile * 16;

    const int lhalf = (lane >= 16) ? 1 : 0;
    const int l15   = lane & 15;
    const int n     = n_base + l15;             // this lane's hidden unit (B/C col)

    v8f acc[2][4] = {};                         // [m-subtile][gate]

    // ---------------- x part: K = 512 (bf16, pre-converted) -----------------
    const __bf16* xrow0 = xbf + ((size_t)(m_base + l15)      * TT + t) * II;
    const __bf16* xrow1 = xbf + ((size_t)(m_base + 16 + l15) * TT + t) * II;

    // Prefetch next timestep's A rows into cache (overlap with this step's math).
    if (t + 1 < TT) {
#pragma unroll
        for (int p = 0; p < II; p += 64) {      // 128B cachelines of bf16
            __builtin_prefetch(xrow0 + II + p, 0, 0);
            __builtin_prefetch(xrow1 + II + p, 0, 0);
        }
    }

    for (int kk = 0; kk < II / 32; ++kk) {
        const int kb = kk * 32 + lhalf * 8;
        v16bf a0 = frag16(xrow0 + kb, xrow0 + kb + 16);
        v16bf a1 = frag16(xrow1 + kb, xrow1 + kb + 16);
#pragma unroll
        for (int g = 0; g < 4; ++g) {
            const __bf16* wrow = wih + (size_t)(g * HH + n) * II + kk * 32 + lhalf * 8;
            v16bf bfrag = frag16(wrow, wrow + 16);
            acc[0][g] = __builtin_amdgcn_wmma_f32_16x16x32_bf16(
                false, a0, false, bfrag, (short)0, acc[0][g], false, false);
            acc[1][g] = __builtin_amdgcn_wmma_f32_16x16x32_bf16(
                false, a1, false, bfrag, (short)0, acc[1][g], false, false);
        }
    }

    // ---------------- h part: K = 768 (bf16 recurrent state) ----------------
    const __bf16* hrow0 = h_in + (size_t)(m_base + l15) * HH;
    const __bf16* hrow1 = h_in + (size_t)(m_base + 16 + l15) * HH;

    for (int kk = 0; kk < HH / 32; ++kk) {
        const int kb = kk * 32 + lhalf * 8;
        v16bf a0 = frag16(hrow0 + kb, hrow0 + kb + 16);
        v16bf a1 = frag16(hrow1 + kb, hrow1 + kb + 16);
#pragma unroll
        for (int g = 0; g < 4; ++g) {
            const __bf16* wrow = whh + (size_t)(g * HH + n) * HH + kk * 32 + lhalf * 8;
            v16bf bfrag = frag16(wrow, wrow + 16);
            acc[0][g] = __builtin_amdgcn_wmma_f32_16x16x32_bf16(
                false, a0, false, bfrag, (short)0, acc[0][g], false, false);
            acc[1][g] = __builtin_amdgcn_wmma_f32_16x16x32_bf16(
                false, a1, false, bfrag, (short)0, acc[1][g], false, false);
        }
    }

    // ---------------- epilogue: gates, c/h update, masking ------------------
    float bias[4];
#pragma unroll
    for (int g = 0; g < 4; ++g)
        bias[g] = bih[g * HH + n] + bhh[g * HH + n];

#pragma unroll
    for (int ms = 0; ms < 2; ++ms) {
        const int mrow0 = m_base + ms * 16 + lhalf * 8;  // C layout: M = r + 8*half
#pragma unroll
        for (int r = 0; r < 8; ++r) {
            const int row = mrow0 + r;                   // batch index
            float ig = fast_sigmoid(acc[ms][0][r] + bias[0]);
            float fg = fast_sigmoid(acc[ms][1][r] + bias[1]);
            float gg = fast_tanh   (acc[ms][2][r] + bias[2]);
            float og = fast_sigmoid(acc[ms][3][r] + bias[3]);

            const size_t ci = (size_t)row * HH + n;
            float cn = fg * c[ci] + ig * gg;
            float hn = og * fast_tanh(cn);

            out[((size_t)row * TT + t) * HH + n] = hn;   // emit BEFORE reset

            float keep = term[row * TT + t] ? 0.0f : 1.0f;
            float hk = hn * keep;
            float ck = cn * keep;
            c[ci]     = ck;
            h_out[ci] = (__bf16)hk;
            if (t == TT - 1) { hf[ci] = hk; cf[ci] = ck; }
        }
    }
}

extern "C" void kernel_launch(void* const* d_in, const int* in_sizes, int n_in,
                              void* d_out, int out_size, void* d_ws, size_t ws_size,
                              hipStream_t stream) {
    const float*         states = (const float*)d_in[0];
    const unsigned char* term   = (const unsigned char*)d_in[1];
    const float*         h0     = (const float*)d_in[2];
    const float*         c0     = (const float*)d_in[3];
    const float*         wih    = (const float*)d_in[4];
    const float*         whh    = (const float*)d_in[5];
    const float*         bih    = (const float*)d_in[6];
    const float*         bhh    = (const float*)d_in[7];
    (void)in_sizes; (void)n_in; (void)out_size; (void)ws_size;

    float* out = (float*)d_out;
    float* hf  = out + (size_t)BB * TT * HH;   // + 25,165,824
    float* cf  = hf + (size_t)BB * HH;         // + 196,608

    // Workspace layout (bytes):
    //   [0,         33554432)  x bf16      (32768*512)
    //   [33554432,  36700160)  w_ih bf16   (3072*512)
    //   [36700160,  41418752)  w_hh bf16   (3072*768)
    //   [41418752,  41811968)  h bf16 buf0 (256*768)
    //   [41811968,  42205184)  h bf16 buf1
    //   [42205184,  42991616)  c fp32      (256*768)
    char*   ws     = (char*)d_ws;
    __bf16* xbf    = (__bf16*)(ws);
    __bf16* wih_bf = (__bf16*)(ws + 33554432);
    __bf16* whh_bf = (__bf16*)(ws + 36700160);
    __bf16* hbf0   = (__bf16*)(ws + 41418752);
    __bf16* hbf1   = (__bf16*)(ws + 41811968);
    float*  cws    = (float*)(ws + 42205184);

    const int nx   = BB * TT * II;  // 16,777,216
    const int nwih = 4 * HH * II;   //  1,572,864
    const int nwhh = 4 * HH * HH;   //  2,359,296
    cvt_f32_bf16_v8<<<(nx / 8 + 255) / 256, 256, 0, stream>>>(states, xbf, nx / 8);
    cvt_f32_bf16_v8<<<(nwih / 8 + 255) / 256, 256, 0, stream>>>(wih, wih_bf, nwih / 8);
    cvt_f32_bf16_v8<<<(nwhh / 8 + 255) / 256, 256, 0, stream>>>(whh, whh_bf, nwhh / 8);
    init_state<<<(BB * HH + 255) / 256, 256, 0, stream>>>(h0, c0, hbf0, cws);

    __bf16* hb[2] = { hbf0, hbf1 };
    for (int t = 0; t < TT; ++t) {
        lstm_step<<<48, 256, 0, stream>>>(xbf, term, wih_bf, whh_bf, bih, bhh,
                                          hb[t & 1], hb[(t + 1) & 1], cws,
                                          out, hf, cf, t);
    }
}
